// GCN_10960756539504
// MI455X (gfx1250) — matrix operand order repaired and checked by reading, not compiled, not run on hardware
//
#include <hip/hip_runtime.h>
#include <hip/hip_bf16.h>
#include <math.h>

// GCN on MI455X (gfx1250): bf16 WMMA GEMMs + L2-resident atomic scatter-add.
// N=100000, F_IN=1433, H=16, C=7, E=3.2M (+N self loops).

typedef __attribute__((ext_vector_type(16))) __bf16 v16bf;
typedef __attribute__((ext_vector_type(8)))  float  v8f;

#define GN   100000
#define GFIN 1433
#define GH   16
#define GC   7
#define KPAD_MAX 1440   // ceil32(1433)

// ---------------- degree / normalization ----------------

__global__ void init_deg(float* deg, int n) {
    int i = blockIdx.x * blockDim.x + threadIdx.x;
    if (i < n) deg[i] = 1.0f;                      // self loop contributes 1
}

__global__ void accum_deg(const long long* __restrict__ dst, float* deg, int e) {
    int i = blockIdx.x * blockDim.x + threadIdx.x;
    if (i < e) atomicAdd(&deg[dst[i]], 1.0f);
}

__global__ void rsqrt_ip(float* d, int n) {
    int i = blockIdx.x * blockDim.x + threadIdx.x;
    if (i < n) d[i] = rsqrtf(d[i]);                // deg >= 1 always
}

// ---------------- GEMM: Y[N,OUTW] = X[N,K] @ W[K,NC], bf16 WMMA ----------------
// One wave32 per 16-row tile, full K loop, 16x16 f32 accumulator.
// W staged transposed in LDS (sW[n*Kpad + k]) as bf16, zero-padded.

// A fragment (16x32 bf16, contiguous-K order, k = 16*hh + s): lane = 16*hh + m;
// hh=0: K = kb..kb+15 in v0..15 (kb = k0)
// hh=1: same with kb = k0+16
__device__ __forceinline__ v16bf load_a_frag(const float* __restrict__ xrow, int kb) {
    v16bf a;
    #pragma unroll
    for (int j = 0; j < 8; ++j) {
        a[j]     = (__bf16)xrow[kb + j];
        a[j + 8] = (__bf16)xrow[kb + 8 + j];
    }
    return a;
}

__global__ __launch_bounds__(256)
void gemm_bf16_wmma(const float* __restrict__ X, const float* __restrict__ W,
                    float* __restrict__ Y, int nrows, int K, int NC, int OUTW)
{
    __shared__ __bf16 sW[16 * KPAD_MAX];
    const int Kpad = (K + 31) & ~31;

    // cooperative fill: transpose + f32->bf16 convert + zero pad
    for (int i = threadIdx.x; i < 16 * Kpad; i += blockDim.x) {
        int n = i / Kpad;
        int k = i - n * Kpad;
        float w = (k < K && n < NC) ? W[k * NC + n] : 0.0f;
        sW[i] = (__bf16)w;
    }
    __syncthreads();

    const int wave = threadIdx.x >> 5;
    const int lane = threadIdx.x & 31;
    const int tile = blockIdx.x * (blockDim.x >> 5) + wave;
    const int ntiles = nrows >> 4;
    if (tile >= ntiles) return;                    // wave-uniform exit

    const int m  = lane & 15;                      // A row within tile
    const int hh = lane >> 4;                      // half-wave select
    const int n  = lane & 15;                      // B/D column
    const float* xrow = X + (size_t)(tile * 16 + m) * (size_t)K;
    const __bf16* wrow = sW + n * Kpad + 16 * hh;  // B: contiguous-K, 32B aligned

    const int Kmain = K & ~31;                     // check-free region

    v8f acc0 = {};
    v8f acc1 = {};
    int k0 = 0;

    // main loop: unroll x2, two accumulators -> independent WMMA chains,
    // next step's loads/converts fill the WMMA hazard shadow.
    for (; k0 + 64 <= Kmain; k0 += 64) {
        v16bf a0 = load_a_frag(xrow, k0 + 16 * hh);
        v16bf b0 = *(const v16bf*)(wrow + k0);
        v16bf a1 = load_a_frag(xrow, k0 + 32 + 16 * hh);
        v16bf b1 = *(const v16bf*)(wrow + k0 + 32);
        acc0 = __builtin_amdgcn_wmma_f32_16x16x32_bf16(
                   false, a0, false, b0, (short)0, acc0, false, false);
        acc1 = __builtin_amdgcn_wmma_f32_16x16x32_bf16(
                   false, a1, false, b1, (short)0, acc1, false, false);
    }
    for (; k0 < Kmain; k0 += 32) {
        v16bf a0 = load_a_frag(xrow, k0 + 16 * hh);
        v16bf b0 = *(const v16bf*)(wrow + k0);
        acc0 = __builtin_amdgcn_wmma_f32_16x16x32_bf16(
                   false, a0, false, b0, (short)0, acc0, false, false);
    }
    // single masked tail step (K not a multiple of 32); B is zero-padded in LDS
    if (Kmain < K) {
        const int kb = Kmain + 16 * hh;
        v16bf a;
        #pragma unroll
        for (int j = 0; j < 8; ++j) {
            int ka = kb + j, kc = kb + 8 + j;
            a[j]     = (ka < K) ? (__bf16)xrow[ka] : (__bf16)0.0f;
            a[j + 8] = (kc < K) ? (__bf16)xrow[kc] : (__bf16)0.0f;
        }
        v16bf b = *(const v16bf*)(wrow + Kmain);
        acc1 = __builtin_amdgcn_wmma_f32_16x16x32_bf16(
                   false, a, false, b, (short)0, acc1, false, false);
    }

    v8f acc = acc0 + acc1;

    // ---- D store: VGPR r, half hh -> row (r + 8*hh), col n
    if (n < OUTW) {
        float* yout = Y + (size_t)(tile * 16 + 8 * hh) * OUTW + n;
        #pragma unroll
        for (int r = 0; r < 8; ++r)
            yout[r * OUTW] = acc[r];
    }
}

// ---------------- edge aggregation: agg[dst] += t[src] * dinv[src]*dinv[dst] ----

template <int CH>
__global__ void aggregate(const float* __restrict__ t,
                          const long long* __restrict__ si,
                          const long long* __restrict__ di,
                          const float* __restrict__ dinv,
                          float* __restrict__ agg, int E_edges, int n_nodes)
{
    int e = blockIdx.x * blockDim.x + threadIdx.x;
    int total = E_edges + n_nodes;
    if (e >= total) return;
    long long s, d; float nrm;
    if (e < E_edges) {
        s = si[e]; d = di[e];
        nrm = dinv[s] * dinv[d];
    } else {                                       // self loop
        s = e - E_edges; d = s;
        float v = dinv[s];
        nrm = v * v;
    }
    const float* ts = t + (size_t)s * CH;
    float* od = agg + (size_t)d * CH;
    #pragma unroll
    for (int c = 0; c < CH; c += 4) {
        float4 v = *(const float4*)(ts + c);
        atomicAdd(od + c + 0, v.x * nrm);
        atomicAdd(od + c + 1, v.y * nrm);
        atomicAdd(od + c + 2, v.z * nrm);
        atomicAdd(od + c + 3, v.w * nrm);
    }
}

__global__ void bias_relu16(float* __restrict__ h, const float* __restrict__ b, int total) {
    int i = blockIdx.x * blockDim.x + threadIdx.x;
    if (i >= total) return;
    float v = h[i] + b[i & 15];
    h[i] = v > 0.0f ? v : 0.0f;
}

__global__ void logsoftmax7(const float* __restrict__ agg, const float* __restrict__ b3,
                            float* __restrict__ out, int n)
{
    int i = blockIdx.x * blockDim.x + threadIdx.x;
    if (i >= n) return;
    float v[GC];
    float m = -1e30f;
    #pragma unroll
    for (int c = 0; c < GC; ++c) { v[c] = agg[i * 8 + c] + b3[c]; m = fmaxf(m, v[c]); }
    float s = 0.0f;
    #pragma unroll
    for (int c = 0; c < GC; ++c) s += __expf(v[c] - m);
    float l = __logf(s);
    #pragma unroll
    for (int c = 0; c < GC; ++c) out[i * GC + c] = v[c] - m - l;
}

// ---------------- orchestration ----------------

extern "C" void kernel_launch(void* const* d_in, const int* in_sizes, int n_in,
                              void* d_out, int out_size, void* d_ws, size_t ws_size,
                              hipStream_t stream)
{
    const float* x  = (const float*)d_in[0];
    const float* W1 = (const float*)d_in[1];
    const float* b1 = (const float*)d_in[2];
    const float* W2 = (const float*)d_in[3];
    const float* b2 = (const float*)d_in[4];
    const float* W3 = (const float*)d_in[5];
    const float* b3 = (const float*)d_in[6];
    const long long* ei = (const long long*)d_in[7];
    const int E_edges = in_sizes[7] / 2;
    const long long* src = ei;
    const long long* dst = ei + E_edges;

    float* dinv = (float*)d_ws;                    // [N]
    float* bufA = dinv + GN;                       // [N,16] gemm out
    float* bufB = bufA + (size_t)GN * 16;          // [N,16] agg / hidden
    float* bufC = bufB + (size_t)GN * 16;          // [N,8]  layer3 gemm out
    float* bufD = bufC + (size_t)GN * 8;           // [N,8]  layer3 agg

    const int B = 256;
    const int ntiles  = GN / 16;                   // 6250, exact
    const int gblocks = (ntiles + 7) / 8;          // 8 waves per block
    const int etot    = E_edges + GN;

    // symmetric normalization (shared by all layers)
    init_deg<<<(GN + B - 1) / B, B, 0, stream>>>(dinv, GN);
    accum_deg<<<(E_edges + B - 1) / B, B, 0, stream>>>(dst, dinv, E_edges);
    rsqrt_ip<<<(GN + B - 1) / B, B, 0, stream>>>(dinv, GN);

    // layer 1: relu(A_hat @ (X@W1) + b1)
    gemm_bf16_wmma<<<gblocks, B, 0, stream>>>(x, W1, bufA, GN, GFIN, GH, GH);
    hipMemsetAsync(bufB, 0, (size_t)GN * 16 * sizeof(float), stream);
    aggregate<16><<<(etot + B - 1) / B, B, 0, stream>>>(bufA, src, dst, dinv, bufB, E_edges, GN);
    bias_relu16<<<(GN * 16 + B - 1) / B, B, 0, stream>>>(bufB, b1, GN * 16);

    // layer 2
    gemm_bf16_wmma<<<gblocks, B, 0, stream>>>(bufB, W2, bufA, GN, GH, GH, GH);
    hipMemsetAsync(bufB, 0, (size_t)GN * 16 * sizeof(float), stream);
    aggregate<16><<<(etot + B - 1) / B, B, 0, stream>>>(bufA, src, dst, dinv, bufB, E_edges, GN);
    bias_relu16<<<(GN * 16 + B - 1) / B, B, 0, stream>>>(bufB, b2, GN * 16);

    // layer 3 + log_softmax (C=7 padded to 8 for aligned float4 aggregation)
    gemm_bf16_wmma<<<gblocks, B, 0, stream>>>(bufB, W3, bufC, GN, GH, GC, 8);
    hipMemsetAsync(bufD, 0, (size_t)GN * 8 * sizeof(float), stream);
    aggregate<8><<<(etot + B - 1) / B, B, 0, stream>>>(bufC, src, dst, dinv, bufD, E_edges, GN);
    logsoftmax7<<<(GN + B - 1) / B, B, 0, stream>>>(bufD, b3, (float*)d_out, GN);
}
